// FlashAttention_46127948759424
// MI455X (gfx1250) — matrix operand order repaired and checked by reading, not compiled, not run on hardware
//
#include <hip/hip_runtime.h>
#include <hip/hip_bf16.h>

typedef __bf16 bf16;
typedef __attribute__((ext_vector_type(8)))  __bf16 bf16x8;
typedef __attribute__((ext_vector_type(16))) __bf16 bf16x16;
typedef __attribute__((ext_vector_type(8)))  float  f32x8;

// Problem constants (from reference)
constexpr int Bb = 2, Tt = 2048, Cc = 1024, Hh = 16, Dd = 64;
constexpr int Mrows = Bb * Tt;                 // 4096
constexpr float SCALE = 0.125f;                // D^-0.5 (power of two: exact in bf16)
constexpr size_t BHTD = (size_t)Bb * Hh * Tt * Dd;  // 4,194,304 elems

__device__ __forceinline__ int lane_id() { return threadIdx.x & 31; }

// ---------------- CDNA5 async global->LDS copy (ASYNCcnt-tracked) -----------
__device__ __forceinline__ void async_copy_b128(void* lds_dst, const void* gsrc) {
  const unsigned lds = (unsigned)(uintptr_t)lds_dst;       // LDS byte address
  const unsigned long long ga = (unsigned long long)(uintptr_t)gsrc;
  asm volatile("global_load_async_to_lds_b128 %0, %1, off"
               :: "v"(lds), "v"(ga) : "memory");
}
__device__ __forceinline__ void wait_async0() {
  asm volatile("s_wait_asynccnt 0" ::: "memory");
}

// ---------------- WMMA fragment helpers (wave32, 16x16x32 bf16) -------------
__device__ __forceinline__ bf16x16 cat8(bf16x8 lo, bf16x8 hi) {
  return __builtin_shufflevector(lo, hi, 0, 1, 2, 3, 4, 5, 6, 7,
                                 8, 9, 10, 11, 12, 13, 14, 15);
}

// A 16x32: lane l holds row m=l&15; K runs kg+0..7 and kg+16..23 (kg = 8*(l>>4)).
__device__ __forceinline__ bf16x16 load_a_frag(const bf16* src, int ld) {
  const int l = lane_id();
  const int m = l & 15, kg = (l >> 4) << 3;
  const bf16* p = src + m * ld + kg;
  return cat8(*(const bf16x8*)(p), *(const bf16x8*)(p + 16));
}

// B 32x16 with B[k][n] = src[k*ld + n]; lane holds column n=l&15; K runs kg+0..7 and kg+16..23 (kg = 8*(l>>4)).
__device__ __forceinline__ bf16x16 load_b_frag(const bf16* src, int ld) {
  const int l = lane_id();
  const int n = l & 15, kg = (l >> 4) << 3;
  bf16x16 r;
#pragma unroll
  for (int s = 0; s < 8; ++s) {
    r[s] = src[(kg + s) * ld + n];
    r[8 + s] = src[(16 + kg + s) * ld + n];
  }
  return r;
}

__device__ __forceinline__ f32x8 wmma_bf16(bf16x16 a, bf16x16 b, f32x8 c) {
  return __builtin_amdgcn_wmma_f32_16x16x32_bf16(false, a, false, b, (short)0, c,
                                                 false, false);
}

__device__ __forceinline__ f32x8 zero8() {
  f32x8 z = {0.f, 0.f, 0.f, 0.f, 0.f, 0.f, 0.f, 0.f};
  return z;
}

// ---------------------------- fp32 -> bf16 convert ---------------------------
// 8 elements per thread: 2x b128 load -> 1x b128 store.
__global__ void convert_f32_bf16(const float* __restrict__ src,
                                 bf16* __restrict__ dst, int n8) {
  int i = blockIdx.x * blockDim.x + threadIdx.x;
  if (i < n8) {
    f32x8 v = ((const f32x8*)src)[i];
    bf16x8 o;
#pragma unroll
    for (int j = 0; j < 8; ++j) o[j] = (bf16)v[j];
    *(bf16x8*)(dst + (size_t)i * 8) = o;
  }
}

// ------------------------------- GEMM kernel ---------------------------------
// C = A(MxK) * B(KxN) + bias.  Block tile 64x256, 8 waves (2x4), wave tile 32x64.
// Double-buffered async staging: tile k+1 is in flight while tile k computes.
// MODE 0: float row-major out.
// MODE 1: bf16 scatter into qkv [3][B][H][T][D], with Q pre-scaled by SCALE.
template <int MODE>
__global__ __launch_bounds__(256) void gemm_bf16(
    const bf16* __restrict__ A, const bf16* __restrict__ Bm,
    const float* __restrict__ bias, float* __restrict__ outF,
    bf16* __restrict__ outQKV, int M, int N, int K) {
  constexpr int LDA = 40, LDB = 264;            // multiples of 8 (16B aligned rows)
  __shared__ bf16 sA[2][64 * LDA];
  __shared__ bf16 sB[2][32 * LDB];

  const int tid = threadIdx.x;
  const int w = tid >> 5, wm = w >> 2, wn = w & 3;
  const int rowBase = blockIdx.x * 64, colBase = blockIdx.y * 256;

  // per-thread staging coordinates (fixed)
  const int ra = tid >> 2, ca = (tid & 3) * 8;

  f32x8 acc[2][4];
#pragma unroll
  for (int i = 0; i < 2; ++i)
#pragma unroll
    for (int j = 0; j < 4; ++j) acc[i][j] = zero8();

  auto issue = [&](int k0, int buf) {
    async_copy_b128(&sA[buf][ra * LDA + ca],
                    &A[(size_t)(rowBase + ra) * K + k0 + ca]);
#pragma unroll
    for (int i = 0; i < 4; ++i) {
      const int ci = tid + i * 256;
      const int rb = ci >> 5, cb = (ci & 31) * 8;
      async_copy_b128(&sB[buf][rb * LDB + cb],
                      &Bm[(size_t)(k0 + rb) * N + colBase + cb]);
    }
  };

  issue(0, 0);
  wait_async0();
  __syncthreads();

  for (int k0 = 0; k0 < K; k0 += 32) {
    const int buf = (k0 >> 5) & 1;
    if (k0 + 32 < K) issue(k0 + 32, buf ^ 1);   // overlap with compute below

    bf16x16 af0 = load_a_frag(&sA[buf][(wm * 32) * LDA], LDA);
    bf16x16 af1 = load_a_frag(&sA[buf][(wm * 32 + 16) * LDA], LDA);
#pragma unroll
    for (int j = 0; j < 4; ++j) {
      bf16x16 bfj = load_b_frag(&sB[buf][wn * 64 + j * 16], LDB);
      acc[0][j] = wmma_bf16(af0, bfj, acc[0][j]);
      acc[1][j] = wmma_bf16(af1, bfj, acc[1][j]);
    }
    wait_async0();
    __syncthreads();
  }

  const int l = lane_id();
  const int nc = l & 15, mh = (l >> 4) << 3;
#pragma unroll
  for (int i = 0; i < 2; ++i)
#pragma unroll
    for (int j = 0; j < 4; ++j)
#pragma unroll
      for (int v = 0; v < 8; ++v) {
        const int r = rowBase + wm * 32 + i * 16 + v + mh;
        const int c = colBase + wn * 64 + j * 16 + nc;
        float val = acc[i][j][v] + bias[c];
        if (MODE == 0) {
          outF[(size_t)r * N + c] = val;
        } else {
          const int which = c >> 10, rem = c & 1023;
          const int h = rem >> 6, d = rem & 63;
          const int bb = r >> 11, t = r & (Tt - 1);
          if (which == 0) val *= SCALE;        // pre-scale Q: exact (2^-3)
          outQKV[(size_t)which * BHTD +
                 (((size_t)(bb * Hh + h)) * Tt + t) * Dd + d] = (bf16)val;
        }
      }
}

// ------------------------------ Flash attention ------------------------------
// One block = (b,h) x 64 query rows; 4 waves, 16 rows each, full D=64 per wave.
// Double-buffered tiles: V via async b128, K held in regs across compute and
// scattered transposed to LDS afterwards. rowsum(P) via WMMA against ones.
__global__ __launch_bounds__(128) void flash_attn(const bf16* __restrict__ qkv,
                                                  bf16* __restrict__ Obf) {
  constexpr int LDKT = 40;   // K^T tile: [64 d][32 kv + pad]
  constexpr int LDV = 72;    // V tile:  [32 kv][64 d + pad]
  constexpr int LDP = 40;    // per-wave P staging: [16 q][32 kv + pad]
  __shared__ bf16 sKT[2][64 * LDKT];
  __shared__ bf16 sV[2][32 * LDV];
  __shared__ bf16 sP[4 * 16 * LDP];

  const int tid = threadIdx.x, w = tid >> 5, l = tid & 31;
  const int bh = blockIdx.x;
  const int b = bh / Hh, h = bh % Hh;
  const int q0b = blockIdx.y * 64, q0w = q0b + w * 16;

  const bf16* Q  = qkv + (size_t)bh * Tt * Dd;
  const bf16* Kp = qkv + BHTD + (size_t)bh * Tt * Dd;
  const bf16* Vp = qkv + 2 * BHTD + (size_t)bh * Tt * Dd;

  bf16x16 aq0 = load_a_frag(Q + (size_t)q0w * Dd, Dd);
  bf16x16 aq1 = load_a_frag(Q + (size_t)q0w * Dd + 32, Dd);

  bf16x16 onesf;
#pragma unroll
  for (int i = 0; i < 16; ++i) onesf[i] = (bf16)1.0f;

  f32x8 o[4];
#pragma unroll
  for (int j = 0; j < 4; ++j) o[j] = zero8();
  f32x8 m_s, l_s;
#pragma unroll
  for (int v = 0; v < 8; ++v) { m_s[v] = -1e30f; l_s[v] = 0.f; }

  const int nc = l & 15, mh = (l >> 4) << 3;
  bf16* sPw = &sP[w * 16 * LDP];
  const int nTiles = (q0b + 64) / 32;  // causal bound for this block

  // fixed per-thread staging coordinates (2 chunks of 8 bf16 per 32x64 tile)
  const int r0 = tid >> 3, c0 = (tid & 7) * 8;
  const int r1 = (tid + 128) >> 3, c1 = (tid & 7) * 8;

  bf16x8 kk0, kk1;  // K chunks in flight (scattered to LDS after compute)
  auto issue_tile = [&](int kv0, int buf) {
    async_copy_b128(&sV[buf][r0 * LDV + c0], &Vp[(size_t)(kv0 + r0) * Dd + c0]);
    async_copy_b128(&sV[buf][r1 * LDV + c1], &Vp[(size_t)(kv0 + r1) * Dd + c1]);
    kk0 = *(const bf16x8*)(&Kp[(size_t)(kv0 + r0) * Dd + c0]);
    kk1 = *(const bf16x8*)(&Kp[(size_t)(kv0 + r1) * Dd + c1]);
  };
  auto scatter_k = [&](int buf) {
#pragma unroll
    for (int jj = 0; jj < 8; ++jj) {
      sKT[buf][(c0 + jj) * LDKT + r0] = kk0[jj];
      sKT[buf][(c1 + jj) * LDKT + r1] = kk1[jj];
    }
  };

  issue_tile(0, 0);
  scatter_k(0);
  wait_async0();
  __syncthreads();

  for (int jt = 0; jt < nTiles; ++jt) {
    const int kv0 = jt * 32;
    const int buf = jt & 1;
    const bool hasNext = (jt + 1 < nTiles);
    if (hasNext) issue_tile(kv0 + 32, buf ^ 1);  // overlaps compute below

    if (q0w + 15 >= kv0) {               // wave-uniform: skip fully masked tiles
      f32x8 s[2];
      s[0] = zero8(); s[1] = zero8();
      // S = (Q*SCALE) * K^T : B[k=d][n=kv] = sKT[d*LDKT + kv]
      s[0] = wmma_bf16(aq0, load_b_frag(&sKT[buf][0], LDKT), s[0]);
      s[0] = wmma_bf16(aq1, load_b_frag(&sKT[buf][32 * LDKT], LDKT), s[0]);
      s[1] = wmma_bf16(aq0, load_b_frag(&sKT[buf][16], LDKT), s[1]);
      s[1] = wmma_bf16(aq1, load_b_frag(&sKT[buf][32 * LDKT + 16], LDKT), s[1]);

      if (kv0 + 31 > q0w) {              // boundary tile only: causal mask
#pragma unroll
        for (int half = 0; half < 2; ++half)
#pragma unroll
          for (int v = 0; v < 8; ++v) {
            const int q = q0w + v + mh;
            const int kv = kv0 + half * 16 + nc;
            if (kv > q) s[half][v] = -1e38f;
          }
      }

      // row max over 32 cols (columns live in the 16-lane halves)
      f32x8 t;
#pragma unroll
      for (int v = 0; v < 8; ++v) t[v] = fmaxf(s[0][v], s[1][v]);
#pragma unroll
      for (int off = 1; off < 16; off <<= 1)
#pragma unroll
        for (int v = 0; v < 8; ++v)
          t[v] = fmaxf(t[v], __shfl_xor(t[v], off, 32));

      f32x8 mnew, alpha;
#pragma unroll
      for (int v = 0; v < 8; ++v) {
        mnew[v] = fmaxf(m_s[v], t[v]);
        alpha[v] = __expf(m_s[v] - mnew[v]);
        m_s[v] = mnew[v];
      }
#pragma unroll
      for (int half = 0; half < 2; ++half)
#pragma unroll
        for (int v = 0; v < 8; ++v)
          s[half][v] = __expf(s[half][v] - mnew[v]);

      // restage P (C-frag layout) into LDS, reload as A-fragment (b128)
#pragma unroll
      for (int half = 0; half < 2; ++half)
#pragma unroll
        for (int v = 0; v < 8; ++v)
          sPw[(v + mh) * LDP + half * 16 + nc] = (bf16)s[half][v];
      bf16x16 ap = load_a_frag(sPw, LDP);

      // rowsum(P) via WMMA against ones: per-row sums land in the same [v]
      // slots as l_s/alpha.
      f32x8 rs = wmma_bf16(ap, onesf, zero8());

#pragma unroll
      for (int v = 0; v < 8; ++v) l_s[v] = l_s[v] * alpha[v] + rs[v];
#pragma unroll
      for (int j = 0; j < 4; ++j)
#pragma unroll
        for (int v = 0; v < 8; ++v) o[j][v] *= alpha[v];

#pragma unroll
      for (int j = 0; j < 4; ++j)
        o[j] = wmma_bf16(ap, load_b_frag(&sV[buf][j * 16], LDV), o[j]);
    }

    if (hasNext) scatter_k(buf ^ 1);     // wait_loadcnt overlapped with WMMAs
    wait_async0();
    __syncthreads();
  }

#pragma unroll
  for (int j = 0; j < 4; ++j)
#pragma unroll
    for (int v = 0; v < 8; ++v) {
      const int t = q0w + v + mh;
      const int col = h * Dd + j * 16 + nc;
      Obf[(size_t)(b * Tt + t) * Cc + col] = (bf16)(o[j][v] / l_s[v]);
    }
}

// --------------------------------- launcher ----------------------------------
extern "C" void kernel_launch(void* const* d_in, const int* in_sizes, int n_in,
                              void* d_out, int out_size, void* d_ws,
                              size_t ws_size, hipStream_t stream) {
  const float* x     = (const float*)d_in[0];
  const float* Wqkv  = (const float*)d_in[1];
  const float* bqkv  = (const float*)d_in[2];
  const float* Wproj = (const float*)d_in[3];
  const float* bproj = (const float*)d_in[4];
  float* out = (float*)d_out;

  // workspace layout (bf16): x | Wqkv | Wproj | qkv[3][B][H][T][D] | O
  bf16* xbf     = (bf16*)d_ws;
  bf16* wqkvbf  = xbf + (size_t)Mrows * Cc;
  bf16* wprojbf = wqkvbf + (size_t)Cc * 3 * Cc;
  bf16* qkvbf   = wprojbf + (size_t)Cc * Cc;
  bf16* obf     = qkvbf + 3 * BHTD;

  const int n1 = Mrows * Cc / 8, n2 = Cc * 3 * Cc / 8, n3 = Cc * Cc / 8;
  convert_f32_bf16<<<(n1 + 255) / 256, 256, 0, stream>>>(x, xbf, n1);
  convert_f32_bf16<<<(n2 + 255) / 256, 256, 0, stream>>>(Wqkv, wqkvbf, n2);
  convert_f32_bf16<<<(n3 + 255) / 256, 256, 0, stream>>>(Wproj, wprojbf, n3);

  dim3 g1(Mrows / 64, (3 * Cc) / 256);
  gemm_bf16<1><<<g1, 256, 0, stream>>>(xbf, wqkvbf, bqkv, nullptr, qkvbf,
                                       Mrows, 3 * Cc, Cc);

  dim3 g2(Bb * Hh, Tt / 64);
  flash_attn<<<g2, 128, 0, stream>>>(qkvbf, obf);

  dim3 g3(Mrows / 64, Cc / 256);
  gemm_bf16<0><<<g3, 256, 0, stream>>>(obf, wprojbf, bproj, out, nullptr,
                                       Mrows, Cc, Cc);
}
